// Qwen3ForCausalLM_56788057587813
// MI455X (gfx1250) — matrix operand order repaired and checked by reading, not compiled, not run on hardware
//
#include <hip/hip_runtime.h>
#include <hip/hip_bf16.h>
#include <math.h>
#include <stdint.h>

// ---------------- model constants ----------------
#define LAYERS 8
#define H      1024
#define NH     16
#define NKV    8
#define HD     128
#define FFI    3072
#define VOCAB  32000
#define BATCH  2
#define SEQ    1024
#define BS     (BATCH*SEQ)      // 2048 tokens
#define EPSV   1e-6f

// ---------------- WMMA types ----------------
typedef __attribute__((ext_vector_type(16))) _Float16 v16h;
typedef __attribute__((ext_vector_type(8)))  float    v8f;

// ---------------- CDNA5 async global->LDS helpers ----------------
// ASYNCcnt-tracked DMA into LDS (CDNA5 ISA).
// lds_off = low 32 bits of the generic address of a __shared__ location.
__device__ __forceinline__ void async_ld_b128(uint32_t lds_off, const float* g) {
    asm volatile("global_load_async_to_lds_b128 %0, %1, off"
                 :: "v"(lds_off), "v"((unsigned long long)(uintptr_t)g) : "memory");
}
__device__ __forceinline__ void async_ld_b32(uint32_t lds_off, const float* g) {
    asm volatile("global_load_async_to_lds_b32 %0, %1, off"
                 :: "v"(lds_off), "v"((unsigned long long)(uintptr_t)g) : "memory");
}
__device__ __forceinline__ void wait_async0() {
    asm volatile("s_wait_asynccnt 0x0" ::: "memory");
}
__device__ __forceinline__ uint32_t lds_offset_of(const void* p) {
    return (uint32_t)(uintptr_t)p;   // LDS generic addr: [31:0] = LDS offset
}

// ---------------- generic batched WMMA GEMM ----------------
// C[z] = alpha * A[z] x B[z] (+ C if accum). B optionally transposed (NT).
// z -> (bb, hh); B uses head hh/kvDiv (GQA).
#define BM 128
#define BN 64
#define BK 32

__global__ __launch_bounds__(256) void gemm_wmma_f16(
    const float* __restrict__ Ag, const float* __restrict__ Bg, float* __restrict__ Cg,
    int M, int N, int K, int lda, int ldb, int ldc,
    long long aB, long long aH, long long bB, long long bH, long long cB, long long cH,
    int nh, int kvDiv, int bTrans, int accum, float alpha)
{
    __shared__ __align__(32) float As[BM * BK];   // [m][k] fp32
    __shared__ __align__(32) float Bs[BN * BK];   // [n][k] fp32

    const int z  = blockIdx.z;
    const int bb = z / nh, hh = z % nh;
    const float* A = Ag + bb * aB + hh * aH;
    const float* B = Bg + bb * bB + (hh / kvDiv) * bH;
    float*       C = Cg + bb * cB + hh * cH;

    const int m0 = blockIdx.y * BM;
    const int n0 = blockIdx.x * BN;

    const int t    = threadIdx.x;      // 0..255, 8 waves of 32
    const int wave = t >> 5;
    const int lane = t & 31;

    // wave tile: 32x32 -> 4 accumulators (2 A frags x 2 B frags)
    const int wm = (wave & 3) * 32;    // row offset within tile
    const int wn = (wave >> 2) * 32;   // col offset within tile

    const uint32_t as_base = lds_offset_of(As);
    const uint32_t bs_base = lds_offset_of(Bs);

    // staging indices (per thread)
    const int ar = t >> 1;             // A row 0..127
    const int ac = (t & 1) * 16;       // A col base {0,16}
    const int br = t >> 2;             // B-NT row 0..63
    const int bc = (t & 3) * 8;        // B-NT col base
    const int kk = t >> 5;             // B-NN k 0..7 step? (see below)

    v8f c00 = {}; v8f c01 = {}; v8f c10 = {}; v8f c11 = {};

    for (int kb = 0; kb < K; kb += BK) {
        // ---- async stage A tile (128x32 fp32): 16 floats/thread, 4x b128 ----
        {
            const float* gsrc = A + (long long)(m0 + ar) * lda + kb + ac;
            uint32_t ldst = as_base + (uint32_t)(ar * BK + ac) * 4u;
            #pragma unroll
            for (int j = 0; j < 4; ++j)
                async_ld_b128(ldst + 16u * j, gsrc + 4 * j);
        }
        // ---- async stage B tile as [n][k] ----
        if (bTrans) {                  // B row-major [N][K]: contiguous b128
            const float* gsrc = B + (long long)(n0 + br) * ldb + kb + bc;
            uint32_t ldst = bs_base + (uint32_t)(br * BK + bc) * 4u;
            async_ld_b128(ldst,       gsrc);
            async_ld_b128(ldst + 16u, gsrc + 4);
        } else {                       // B row-major [K][N]: dword scatter-transpose
            const int i0 = t * 8;
            const int k2 = i0 >> 6;    // 0..31
            const int nn = i0 & 63;    // 8-aligned
            const float* gsrc = B + (long long)(kb + k2) * ldb + n0 + nn;
            #pragma unroll
            for (int j = 0; j < 8; ++j)
                async_ld_b32(bs_base + (uint32_t)((nn + j) * BK + k2) * 4u, gsrc + j);
        }
        wait_async0();
        __syncthreads();

        // ---- build fragments (fp32 LDS -> f16 regs) ----
        // A 16x32 layout: lanes 0-15: M=lane, K 0..7 / 16..23; lanes 16-31: K 8..15 / 24..31
        const int kgrp = (lane >> 4) * 8;
        v16h a0, a1;
        {
            const int am = wm + (lane & 15);
            v8f lo = *(const v8f*)&As[am * BK + kgrp];
            v8f hi = *(const v8f*)&As[am * BK + 16 + kgrp];
            #pragma unroll
            for (int j = 0; j < 8; ++j) { a0[j] = (_Float16)lo[j]; a0[8 + j] = (_Float16)hi[j]; }
        }
        {
            const int am = wm + 16 + (lane & 15);
            v8f lo = *(const v8f*)&As[am * BK + kgrp];
            v8f hi = *(const v8f*)&As[am * BK + 16 + kgrp];
            #pragma unroll
            for (int j = 0; j < 8; ++j) { a1[j] = (_Float16)lo[j]; a1[8 + j] = (_Float16)hi[j]; }
        }
        // B 32x16 layout: lanes 0-15: N=lane, K 0..7 / 16..23; lanes 16-31: N=lane-16, K 8..15 / 24..31
        const int kb2 = (lane >> 4) * 8;
        v16h b0, b1;
        {
            const int bn = wn + (lane & 15);
            v8f lo = *(const v8f*)&Bs[bn * BK + kb2];
            v8f hi = *(const v8f*)&Bs[bn * BK + kb2 + 16];
            #pragma unroll
            for (int j = 0; j < 8; ++j) { b0[j] = (_Float16)lo[j]; b0[8 + j] = (_Float16)hi[j]; }
        }
        {
            const int bn = wn + 16 + (lane & 15);
            v8f lo = *(const v8f*)&Bs[bn * BK + kb2];
            v8f hi = *(const v8f*)&Bs[bn * BK + kb2 + 16];
            #pragma unroll
            for (int j = 0; j < 8; ++j) { b1[j] = (_Float16)lo[j]; b1[8 + j] = (_Float16)hi[j]; }
        }

        c00 = __builtin_amdgcn_wmma_f32_16x16x32_f16(false, a0, false, b0, (short)0, c00, false, false);
        c01 = __builtin_amdgcn_wmma_f32_16x16x32_f16(false, a0, false, b1, (short)0, c01, false, false);
        c10 = __builtin_amdgcn_wmma_f32_16x16x32_f16(false, a1, false, b0, (short)0, c10, false, false);
        c11 = __builtin_amdgcn_wmma_f32_16x16x32_f16(false, a1, false, b1, (short)0, c11, false, false);

        __syncthreads();
    }

    // ---- write C: lanes 0-15 rows 0..7, lanes 16-31 rows 8..15 of each 16x16 ----
    const int rl = (lane >> 4) ? 8 : 0;
    const int cl = lane & 15;
    #pragma unroll
    for (int i = 0; i < 2; ++i) {
        v8f ci0 = i ? c10 : c00;
        v8f ci1 = i ? c11 : c01;
        const int cm = m0 + wm + 16 * i + rl;
        const int cn = n0 + wn + cl;
        #pragma unroll
        for (int r = 0; r < 8; ++r) {
            float* p0 = C + (long long)(cm + r) * ldc + cn;
            float* p1 = p0 + 16;
            float v0 = alpha * ci0[r];
            float v1 = alpha * ci1[r];
            if (accum) { v0 += *p0; v1 += *p1; }
            *p0 = v0; *p1 = v1;
        }
    }
    (void)kk;
}

// ---------------- embedding gather ----------------
__global__ __launch_bounds__(256) void embed_gather_k(
    const int* __restrict__ ids, const float* __restrict__ embed, float* __restrict__ h)
{
    const int tok = blockIdx.x;
    const long long id = ids[tok];
    const float* src = embed + id * (long long)H;
    float* dst = h + (long long)tok * H;
    for (int j = threadIdx.x; j < H; j += 256) dst[j] = src[j];
}

// ---------------- RMSNorm (one block per row) ----------------
__global__ __launch_bounds__(256) void rmsnorm_k(
    const float* __restrict__ in, const float* __restrict__ w,
    float* __restrict__ out, int width)
{
    const long long row = blockIdx.x;
    const float* x = in + row * width;
    float* y = out + row * width;
    const int t = threadIdx.x;
    float ss = 0.f;
    for (int j = t; j < width; j += 256) { float v = x[j]; ss += v * v; }
    __shared__ float red[256];
    red[t] = ss; __syncthreads();
    for (int s = 128; s > 0; s >>= 1) { if (t < s) red[t] += red[t + s]; __syncthreads(); }
    const float inv = rsqrtf(red[0] / (float)width + EPSV);
    for (int j = t; j < width; j += 256) y[j] = x[j] * inv * w[j];
}

// ---------------- RoPE (in place), one thread per rotation pair ----------------
__global__ __launch_bounds__(256) void rope_k(float* __restrict__ x, int nheads)
{
    const long long tid = (long long)blockIdx.x * 256 + threadIdx.x;
    const int i = (int)(tid & 63);                 // pair index 0..63
    const long long r = tid >> 6;                  // ((b*S+s)*nheads + h)
    const int s = (int)((r / nheads) % SEQ);
    float* p = x + r * (long long)HD;
    const float invf = powf(1.0e6f, -(float)i / 64.0f);
    const float f = (float)s * invf;
    float sn, cs;
    sincosf(f, &sn, &cs);
    const float x1 = p[i];
    const float x2 = p[i + 64];
    p[i]      = x1 * cs - x2 * sn;
    p[i + 64] = x2 * cs + x1 * sn;
}

// ---------------- causal softmax over one score row ----------------
__global__ __launch_bounds__(256) void softmax_causal_k(float* __restrict__ scores)
{
    const int q = blockIdx.x;                  // query position
    float* row = scores + ((long long)blockIdx.y * SEQ + q) * SEQ;
    const int t = threadIdx.x;
    const int len = q + 1;
    __shared__ float red[256];

    float mx = -3.4e38f;
    for (int j = t; j < len; j += 256) mx = fmaxf(mx, row[j]);
    red[t] = mx; __syncthreads();
    for (int s = 128; s > 0; s >>= 1) { if (t < s) red[t] = fmaxf(red[t], red[t + s]); __syncthreads(); }
    mx = red[0]; __syncthreads();

    float sum = 0.f;
    for (int j = t; j < len; j += 256) { float e = expf(row[j] - mx); row[j] = e; sum += e; }
    red[t] = sum; __syncthreads();
    for (int s = 128; s > 0; s >>= 1) { if (t < s) red[t] += red[t + s]; __syncthreads(); }
    const float inv = 1.0f / red[0];

    for (int j = t; j < len; j += 256) row[j] *= inv;
    for (int j = len + t; j < SEQ; j += 256) row[j] = 0.f;
}

// ---------------- g = silu(g) * u ----------------
__global__ __launch_bounds__(256) void silu_mul_k(
    float* __restrict__ g, const float* __restrict__ u, long long n)
{
    const long long i = (long long)blockIdx.x * 256 + threadIdx.x;
    if (i < n) {
        const float x = g[i];
        const float s = 1.0f / (1.0f + expf(-x));
        g[i] = x * s * u[i];
    }
}

// ---------------- host-side GEMM launcher ----------------
static inline void launch_gemm(hipStream_t stream,
    const float* A, const float* B, float* C,
    int M, int N, int K, int lda, int ldb, int ldc,
    long long aB, long long aH, long long bB, long long bH, long long cB, long long cH,
    int nh, int kvDiv, int Z, int bTrans, int accum, float alpha)
{
    dim3 grid(N / BN, M / BM, Z), block(256);
    hipLaunchKernelGGL(gemm_wmma_f16, grid, block, 0, stream,
                       A, B, C, M, N, K, lda, ldb, ldc,
                       aB, aH, bB, bH, cB, cH, nh, kvDiv, bTrans, accum, alpha);
}

extern "C" void kernel_launch(void* const* d_in, const int* in_sizes, int n_in,
                              void* d_out, int out_size, void* d_ws, size_t ws_size,
                              hipStream_t stream)
{
    (void)in_sizes; (void)n_in; (void)out_size; (void)ws_size;

    const int*   ids     = (const int*)  d_in[0];
    const float* embed   = (const float*)d_in[1];
    const float* Wq      = (const float*)d_in[2];
    const float* Wk      = (const float*)d_in[3];
    const float* Wv      = (const float*)d_in[4];
    const float* Wo      = (const float*)d_in[5];
    const float* qn      = (const float*)d_in[6];
    const float* kn      = (const float*)d_in[7];
    const float* ln1     = (const float*)d_in[8];
    const float* ln2     = (const float*)d_in[9];
    const float* Wg      = (const float*)d_in[10];
    const float* Wu      = (const float*)d_in[11];
    const float* Wd      = (const float*)d_in[12];
    const float* norm_w  = (const float*)d_in[13];
    const float* lm_head = (const float*)d_in[14];
    float* out = (float*)d_out;

    // ---- workspace layout (floats) ----
    float* ws = (float*)d_ws;
    float* h      = ws;                         // BS*H
    float* x      = h  + (long long)BS * H;     // BS*H
    float* q      = x  + (long long)BS * H;     // BS*NH*HD (also ctx)
    float* k      = q  + (long long)BS * NH * HD;   // BS*NKV*HD
    float* v      = k  + (long long)BS * NKV * HD;  // BS*NKV*HD
    float* g      = v  + (long long)BS * NKV * HD;  // BS*I
    float* u      = g  + (long long)BS * FFI;       // BS*I
    float* scores = u  + (long long)BS * FFI;       // B*NH*S*S

    const float scale = 1.0f / sqrtf((float)HD);

    // ---- embedding ----
    hipLaunchKernelGGL(embed_gather_k, dim3(BS), dim3(256), 0, stream, ids, embed, h);

    for (int l = 0; l < LAYERS; ++l) {
        const float* Wq_l  = Wq  + (long long)l * H * NH * HD;
        const float* Wk_l  = Wk  + (long long)l * H * NKV * HD;
        const float* Wv_l  = Wv  + (long long)l * H * NKV * HD;
        const float* Wo_l  = Wo  + (long long)l * NH * HD * H;
        const float* qn_l  = qn  + (long long)l * HD;
        const float* kn_l  = kn  + (long long)l * HD;
        const float* ln1_l = ln1 + (long long)l * H;
        const float* ln2_l = ln2 + (long long)l * H;
        const float* Wg_l  = Wg  + (long long)l * H * FFI;
        const float* Wu_l  = Wu  + (long long)l * H * FFI;
        const float* Wd_l  = Wd  + (long long)l * FFI * H;

        // x = rmsnorm(h, ln1)
        hipLaunchKernelGGL(rmsnorm_k, dim3(BS), dim3(256), 0, stream, h, ln1_l, x, H);

        // q,k,v projections
        launch_gemm(stream, x, Wq_l, q, BS, NH * HD, H, H, NH * HD, NH * HD,
                    0, 0, 0, 0, 0, 0, 1, 1, 1, 0, 0, 1.0f);
        launch_gemm(stream, x, Wk_l, k, BS, NKV * HD, H, H, NKV * HD, NKV * HD,
                    0, 0, 0, 0, 0, 0, 1, 1, 1, 0, 0, 1.0f);
        launch_gemm(stream, x, Wv_l, v, BS, NKV * HD, H, H, NKV * HD, NKV * HD,
                    0, 0, 0, 0, 0, 0, 1, 1, 1, 0, 0, 1.0f);

        // per-head rmsnorm on q,k (in place)
        hipLaunchKernelGGL(rmsnorm_k, dim3(BS * NH),  dim3(256), 0, stream, q, qn_l, q, HD);
        hipLaunchKernelGGL(rmsnorm_k, dim3(BS * NKV), dim3(256), 0, stream, k, kn_l, k, HD);

        // RoPE (in place)
        hipLaunchKernelGGL(rope_k, dim3((BS * NH  * 64) / 256), dim3(256), 0, stream, q, NH);
        hipLaunchKernelGGL(rope_k, dim3((BS * NKV * 64) / 256), dim3(256), 0, stream, k, NKV);

        // scores = scale * q . k^T   (batched over B*NH heads, GQA kvDiv=2)
        launch_gemm(stream, q, k, scores, SEQ, SEQ, HD,
                    NH * HD, NKV * HD, SEQ,
                    (long long)SEQ * NH * HD, HD,
                    (long long)SEQ * NKV * HD, HD,
                    (long long)NH * SEQ * SEQ, (long long)SEQ * SEQ,
                    NH, NH / NKV, BATCH * NH, /*bTrans=*/1, 0, scale);

        // causal softmax
        hipLaunchKernelGGL(softmax_causal_k, dim3(SEQ, BATCH * NH), dim3(256), 0, stream, scores);

        // ctx = attn . v  (write into q buffer, layout (B,S,NH,HD))
        launch_gemm(stream, scores, v, q, SEQ, HD, SEQ,
                    SEQ, NKV * HD, NH * HD,
                    (long long)NH * SEQ * SEQ, (long long)SEQ * SEQ,
                    (long long)SEQ * NKV * HD, HD,
                    (long long)SEQ * NH * HD, HD,
                    NH, NH / NKV, BATCH * NH, /*bTrans=*/0, 0, 1.0f);

        // h += ctx @ Wo
        launch_gemm(stream, q, Wo_l, h, BS, H, NH * HD, NH * HD, H, H,
                    0, 0, 0, 0, 0, 0, 1, 1, 1, 0, /*accum=*/1, 1.0f);

        // MLP
        hipLaunchKernelGGL(rmsnorm_k, dim3(BS), dim3(256), 0, stream, h, ln2_l, x, H);
        launch_gemm(stream, x, Wg_l, g, BS, FFI, H, H, FFI, FFI,
                    0, 0, 0, 0, 0, 0, 1, 1, 1, 0, 0, 1.0f);
        launch_gemm(stream, x, Wu_l, u, BS, FFI, H, H, FFI, FFI,
                    0, 0, 0, 0, 0, 0, 1, 1, 1, 0, 0, 1.0f);
        {
            const long long n = (long long)BS * FFI;
            hipLaunchKernelGGL(silu_mul_k, dim3((unsigned)((n + 255) / 256)), dim3(256), 0, stream, g, u, n);
        }
        // h += act @ Wd
        launch_gemm(stream, g, Wd_l, h, BS, H, FFI, FFI, H, H,
                    0, 0, 0, 0, 0, 0, 1, 1, 1, 0, /*accum=*/1, 1.0f);
    }

    // final norm + lm_head
    hipLaunchKernelGGL(rmsnorm_k, dim3(BS), dim3(256), 0, stream, h, norm_w, x, H);
    launch_gemm(stream, x, lm_head, out, BS, VOCAB, H, H, VOCAB, VOCAB,
                0, 0, 0, 0, 0, 0, 1, 1, 1, 0, 0, 1.0f);
}
